// EncoderDecoder_21234318311609
// MI455X (gfx1250) — compile-verified, matrix operand order checked by reading, not run on hardware
//
#include <hip/hip_runtime.h>
#include <hip/hip_bf16.h>
#include <math.h>

// ---------------------------------------------------------------------------
// EncoderDecoder for MI455X (gfx1250).
//  * All GEMMs -> v_wmma_f32_16x16x32_bf16 (fp32 accumulate).
//  * LSTM timestep fully fused: one kernel does [x_t | h] @ Wint^T (K=2048,
//    gate-interleaved weights), stages the 32x256 fp32 gate tile in LDS, and
//    applies the sigmoid/tanh cell update in place -> c (fp32), h (bf16).
//    No gates buffer in HBM, 1 launch per serial timestep.
//  * bf16 weight copies (~109MB) are L2-resident (192MB) across the scan.
// ---------------------------------------------------------------------------

typedef __attribute__((ext_vector_type(16))) __bf16 v16bf;
typedef __attribute__((ext_vector_type(8)))  float  v8f;
typedef unsigned short u16;
typedef unsigned int   u32;

constexpr int B_   = 256;
constexpr int T_   = 128;
constexpr int IN_  = 64;
constexpr int H_   = 1024;
constexpr int G_   = 4 * H_;     // 4096 gate width
constexpr int K2_  = 2 * H_;     // 2048 fused K ( [x | h] )
constexpr int QW_  = 128;
constexpr int QL_  = 128;
constexpr int V_   = QW_ * QL_;  // 16384
constexpr int KB_  = 5;
constexpr int DELTA_ = 16;

struct alignas(16) U4 { u32 x[4]; };

__device__ __forceinline__ u16 f32_to_bf16(float f) {
  union { float f; u32 u; } x; x.f = f;
  u32 u = x.u;
  u += 0x7FFFu + ((u >> 16) & 1u);   // round-to-nearest-even
  return (u16)(u >> 16);
}

// 16x32 bf16 tile (row-major, row stride `ld`) -> WMMA fragment layout:
// lanes 0-15 hold row r=lane, K=0..7 & 16..23; lanes 16-31 row r=lane-16,
// K=8..15 & 24..31. Two global_load_b128 per lane.
__device__ __forceinline__ v16bf load_frag16x32(const u16* __restrict__ base, int ld) {
  const int lane = threadIdx.x & 31;
  const int r    = lane & 15;
  const int hh   = lane >> 4;
  const u16* p = base + (size_t)r * ld + hh * 8;
  union { v16bf v; U4 q[2]; } f;
  f.q[0] = *reinterpret_cast<const U4*>(p);
  f.q[1] = *reinterpret_cast<const U4*>(p + 16);
  return f.v;
}

// ---------------------------------------------------------------------------
// Plain GEMM: C[M,N] = A[M,K] @ W[N,K]^T + bias[N].  Out fp32 or bf16.
// Block = 8 waves = 32x256 tile; wave = 16x64 strip, 4 accumulators.
// ---------------------------------------------------------------------------
__global__ __launch_bounds__(256)
void gemm_bf16_wmma(const u16* __restrict__ A, int lda,
                    const u16* __restrict__ W,
                    const float* __restrict__ bias,
                    float* __restrict__ outF, u16* __restrict__ outB,
                    int M, int N, int K)
{
  const int wave = threadIdx.x >> 5;
  const int lane = threadIdx.x & 31;
  const int m0 = blockIdx.y * 32  + (wave >> 2) * 16;
  const int n0 = blockIdx.x * 256 + (wave & 3) * 64;

  v8f acc0 = {}, acc1 = {}, acc2 = {}, acc3 = {};
  const u16* a_row = A + (size_t)m0 * lda;

  for (int k = 0; k < K; k += 32) {
    v16bf a  = load_frag16x32(a_row + k, lda);
    v16bf b0 = load_frag16x32(W + (size_t)(n0 +  0) * K + k, K);
    v16bf b1 = load_frag16x32(W + (size_t)(n0 + 16) * K + k, K);
    v16bf b2 = load_frag16x32(W + (size_t)(n0 + 32) * K + k, K);
    v16bf b3 = load_frag16x32(W + (size_t)(n0 + 48) * K + k, K);
    acc0 = __builtin_amdgcn_wmma_f32_16x16x32_bf16(false, a, false, b0, (short)0, acc0, false, false);
    acc1 = __builtin_amdgcn_wmma_f32_16x16x32_bf16(false, a, false, b1, (short)0, acc1, false, false);
    acc2 = __builtin_amdgcn_wmma_f32_16x16x32_bf16(false, a, false, b2, (short)0, acc2, false, false);
    acc3 = __builtin_amdgcn_wmma_f32_16x16x32_bf16(false, a, false, b3, (short)0, acc3, false, false);
  }

  const int half = lane >> 4;
  const int nl   = lane & 15;
  v8f accs[4] = {acc0, acc1, acc2, acc3};
#pragma unroll
  for (int j = 0; j < 4; ++j) {
    const int col = n0 + j * 16 + nl;
    const float bv = bias ? bias[col] : 0.0f;
#pragma unroll
    for (int i = 0; i < 8; ++i) {
      const int row = m0 + i + 8 * half;          // C/D layout: VGPR i -> M = i + 8*half
      const size_t o = (size_t)row * N + col;
      const float v = accs[j][i] + bv;
      if (outB) outB[o] = f32_to_bf16(v);
      else      outF[o] = v;
    }
  }
}

// ---------------------------------------------------------------------------
// Fused LSTM timestep.
//   gates(32x256 tile of B x 4096) = [x_t | h_prev] @ Wint^T + Bint
// with gate-interleaved columns: col n = 4*u + gate, so a block tile holds
// complete (i,f,g,o) quadruples for 64 hidden units.  The fp32 tile is staged
// in LDS, then the cell update runs in-kernel:
//   c' = sig(f)*c + sig(i)*tanh(g);  h' = sig(o)*tanh(c')
// h ping-pongs (h_out != h_prev) since other blocks may still read h_prev.
// ---------------------------------------------------------------------------
__global__ __launch_bounds__(256)
void lstm_step_fused(const u16* __restrict__ X, int ldx,    // x_t rows (bf16)
                     const u16* __restrict__ Hp,            // h_prev (B x H bf16)
                     const u16* __restrict__ Wint,          // 4096 x 2048 bf16
                     const float* __restrict__ Bint,        // 4096 (bih+bhh, interleaved)
                     float* __restrict__ c,                 // B x H fp32 (in/out)
                     u16* __restrict__ h_out,               // B x H bf16
                     u16* __restrict__ ys, int ldy)         // optional (B,T,H) mirror
{
  __shared__ float tile[32][264];                           // 32x256 + pad

  const int wave = threadIdx.x >> 5;
  const int lane = threadIdx.x & 31;
  const int m0l  = (wave >> 2) * 16;                        // 0 / 16
  const int n0l  = (wave & 3) * 64;
  const int m0   = blockIdx.y * 32  + m0l;
  const int n0   = blockIdx.x * 256 + n0l;

  v8f acc0 = {}, acc1 = {}, acc2 = {}, acc3 = {};
  const u16* x_row = X  + (size_t)m0 * ldx;
  const u16* h_row = Hp + (size_t)m0 * H_;

  for (int k = 0; k < K2_; k += 32) {
    v16bf a = (k < H_) ? load_frag16x32(x_row + k, ldx)
                       : load_frag16x32(h_row + (k - H_), H_);
    v16bf b0 = load_frag16x32(Wint + (size_t)(n0 +  0) * K2_ + k, K2_);
    v16bf b1 = load_frag16x32(Wint + (size_t)(n0 + 16) * K2_ + k, K2_);
    v16bf b2 = load_frag16x32(Wint + (size_t)(n0 + 32) * K2_ + k, K2_);
    v16bf b3 = load_frag16x32(Wint + (size_t)(n0 + 48) * K2_ + k, K2_);
    acc0 = __builtin_amdgcn_wmma_f32_16x16x32_bf16(false, a, false, b0, (short)0, acc0, false, false);
    acc1 = __builtin_amdgcn_wmma_f32_16x16x32_bf16(false, a, false, b1, (short)0, acc1, false, false);
    acc2 = __builtin_amdgcn_wmma_f32_16x16x32_bf16(false, a, false, b2, (short)0, acc2, false, false);
    acc3 = __builtin_amdgcn_wmma_f32_16x16x32_bf16(false, a, false, b3, (short)0, acc3, false, false);
  }

  // Stage gate tile (+ fused bias) into LDS.
  const int half = lane >> 4;
  const int nl   = lane & 15;
  v8f accs[4] = {acc0, acc1, acc2, acc3};
#pragma unroll
  for (int j = 0; j < 4; ++j) {
    const int cl = n0l + j * 16 + nl;
    const float bv = Bint[n0 + j * 16 + nl];
#pragma unroll
    for (int i = 0; i < 8; ++i) {
      tile[m0l + i + 8 * half][cl] = accs[j][i] + bv;
    }
  }
  __syncthreads();

  // Cell update: 32 rows x 64 units per block, 8 (b,u) pairs per thread.
#pragma unroll
  for (int it = threadIdx.x; it < 32 * 64; it += 256) {
    const int rl = it >> 6;          // local batch row
    const int ul = it & 63;          // local hidden unit
    const float gi = tile[rl][ul * 4 + 0];
    const float gf = tile[rl][ul * 4 + 1];
    const float gg = tile[rl][ul * 4 + 2];
    const float go = tile[rl][ul * 4 + 3];
    const int b = blockIdx.y * 32 + rl;
    const int u = blockIdx.x * 64 + ul;
    const size_t o = (size_t)b * H_ + u;
    const float si = 1.0f / (1.0f + expf(-gi));
    const float sf = 1.0f / (1.0f + expf(-gf));
    const float so = 1.0f / (1.0f + expf(-go));
    const float cn = sf * c[o] + si * tanhf(gg);
    const float hn = so * tanhf(cn);
    c[o] = cn;
    const u16 hb = f32_to_bf16(hn);
    h_out[o] = hb;
    if (ys) ys[(size_t)b * ldy + u] = hb;
  }
}

// Build gate-interleaved bf16 weights + fused bias for one LSTM:
//   Wint[(u*4+g), 0:H]  = wih[g*H+u, :],  Wint[(u*4+g), H:2H] = whh[g*H+u, :]
//   Bint[u*4+g] = bih[g*H+u] + bhh[g*H+u]
__global__ void build_lstm_w(const float* __restrict__ wih, const float* __restrict__ whh,
                             const float* __restrict__ bih, const float* __restrict__ bhh,
                             u16* __restrict__ wout, float* __restrict__ bout)
{
  const long long idx = (long long)blockIdx.x * blockDim.x + threadIdx.x;
  if (idx >= (long long)G_ * K2_) return;
  const int n = (int)(idx >> 11);          // 0..4095  (K2_ = 2048)
  const int k = (int)(idx & (K2_ - 1));
  const int u = n >> 2;
  const int g = n & 3;
  const int srow = g * H_ + u;
  const float v = (k < H_) ? wih[(size_t)srow * H_ + k]
                           : whh[(size_t)srow * H_ + (k - H_)];
  wout[idx] = f32_to_bf16(v);
  if (k == 0) bout[n] = bih[srow] + bhh[srow];
}

__global__ void f32_to_bf16_kernel(const float* __restrict__ in, u16* __restrict__ out, int n) {
  int i = blockIdx.x * blockDim.x + threadIdx.x;
  if (i < n) out[i] = f32_to_bf16(in[i]);
}

// One block per batch row: kcur sequential argmax passes over V scores
// (lowest index wins ties, matching lax.top_k), masking winners with -inf.
__global__ __launch_bounds__(256)
void beam_kernel(float* __restrict__ scores,
                 float* __restrict__ decoded,   // (B, DELTA, K, 2) fp32
                 int* __restrict__ top1,
                 int step, int kcur)
{
  __shared__ float smax[256];
  __shared__ int   sidx[256];
  const int b = blockIdx.x;
  float* row = scores + (size_t)b * V_;

  for (int p = 0; p < kcur; ++p) {
    float best = -INFINITY; int bi = V_;
    for (int i = threadIdx.x; i < V_; i += 256) {
      const float v = row[i];
      if (v > best) { best = v; bi = i; }
    }
    smax[threadIdx.x] = best; sidx[threadIdx.x] = bi;
    __syncthreads();
    for (int s = 128; s > 0; s >>= 1) {
      if (threadIdx.x < s) {
        const float vo = smax[threadIdx.x + s];
        const int   io = sidx[threadIdx.x + s];
        if (vo > smax[threadIdx.x] ||
            (vo == smax[threadIdx.x] && io < sidx[threadIdx.x])) {
          smax[threadIdx.x] = vo; sidx[threadIdx.x] = io;
        }
      }
      __syncthreads();
    }
    if (threadIdx.x == 0) {
      const int idx = sidx[0];
      const int fw = idx / QL_;
      const int fl = idx % QL_;
      float* d = decoded + (((size_t)b * DELTA_ + step) * KB_ + p) * 2;
      d[0] = (float)fl;
      d[1] = (float)fw;
      if (p == 0) top1[b] = idx;
      row[idx] = -INFINITY;
    }
    __syncthreads();
  }
}

// emb = concat(fqw_w[:, fw] + fqw_b, fql_w[:, fl] + fql_b)  (one-hot @ W^T)
__global__ void emb_kernel(const int* __restrict__ top1,
                           const float* __restrict__ fqw_w, const float* __restrict__ fqw_b,
                           const float* __restrict__ fql_w, const float* __restrict__ fql_b,
                           u16* __restrict__ emb)
{
  int idx = blockIdx.x * blockDim.x + threadIdx.x;
  if (idx >= B_ * H_) return;
  const int b = idx >> 10;
  const int j = idx & (H_ - 1);
  const int t = top1[b];
  const int fw = t / QL_;
  const int fl = t % QL_;
  float v;
  if (j < H_ / 2) v = fqw_w[(size_t)j * QW_ + fw] + fqw_b[j];
  else {
    const int jj = j - H_ / 2;
    v = fql_w[(size_t)jj * QL_ + fl] + fql_b[jj];
  }
  emb[idx] = f32_to_bf16(v);
}

// ---------------------------------------------------------------------------

extern "C" void kernel_launch(void* const* d_in, const int* in_sizes, int n_in,
                              void* d_out, int out_size, void* d_ws, size_t ws_size,
                              hipStream_t stream) {
  (void)in_sizes; (void)n_in; (void)ws_size;

  char* wsb = (char*)d_ws;
  size_t off = 0;
  auto alloc = [&](size_t bytes) -> void* {
    void* p = wsb + off;
    off = (off + bytes + 255) & ~(size_t)255;
    return p;
  };

  auto conv = [&](int idx, size_t n) -> u16* {
    u16* p = (u16*)alloc(n * sizeof(u16));
    f32_to_bf16_kernel<<<(int)((n + 255) / 256), 256, 0, stream>>>(
        (const float*)d_in[idx], p, (int)n);
    return p;
  };

  // ---- bf16 FC weights ----
  u16* efc1_w = conv(1,  (size_t)H_ * IN_);
  u16* efc2_w = conv(3,  (size_t)H_ * H_);
  u16* efc3_w = conv(5,  (size_t)H_ * H_);
  u16* dfc1_w = conv(23, (size_t)H_ * H_);
  u16* dfc2_w = conv(25, (size_t)H_ * H_);
  u16* dfc3_w = conv(27, (size_t)V_ * H_);

  // ---- gate-interleaved fused LSTM weights ([wih|whh], bih+bhh) ----
  auto lstm_w = [&](int wih, int whh, int bih, int bhh, u16*& wout, float*& bout) {
    wout = (u16*)alloc((size_t)G_ * K2_ * sizeof(u16));
    bout = (float*)alloc((size_t)G_ * sizeof(float));
    const long long n = (long long)G_ * K2_;
    build_lstm_w<<<(int)((n + 255) / 256), 256, 0, stream>>>(
        (const float*)d_in[wih], (const float*)d_in[whh],
        (const float*)d_in[bih], (const float*)d_in[bhh], wout, bout);
  };
  u16 *el1_w, *el2_w, *dl1_w, *dl2_w;
  float *el1_b, *el2_b, *dl1_b, *dl2_b;
  lstm_w( 7,  8,  9, 10, el1_w, el1_b);
  lstm_w(11, 12, 13, 14, el2_w, el2_b);
  lstm_w(15, 16, 17, 18, dl1_w, dl1_b);
  lstm_w(19, 20, 21, 22, dl2_w, dl2_b);

  // ---- activations / state ----
  u16*   x_bf  = conv(0, (size_t)B_ * T_ * IN_);
  u16*   act0  = (u16*)alloc((size_t)B_ * T_ * H_ * sizeof(u16));
  u16*   act1  = (u16*)alloc((size_t)B_ * T_ * H_ * sizeof(u16));
  float* c1    = (float*)alloc((size_t)B_ * H_ * sizeof(float));
  float* c2    = (float*)alloc((size_t)B_ * H_ * sizeof(float));
  u16*   h1a   = (u16*)alloc((size_t)B_ * H_ * sizeof(u16));
  u16*   h1b   = (u16*)alloc((size_t)B_ * H_ * sizeof(u16));
  u16*   h2a   = (u16*)alloc((size_t)B_ * H_ * sizeof(u16));
  u16*   h2b   = (u16*)alloc((size_t)B_ * H_ * sizeof(u16));
  u16*   emb   = (u16*)alloc((size_t)B_ * H_ * sizeof(u16));
  u16*   bufA  = (u16*)alloc((size_t)B_ * H_ * sizeof(u16));
  u16*   bufB  = (u16*)alloc((size_t)B_ * H_ * sizeof(u16));
  float* scores = (float*)alloc((size_t)B_ * V_ * sizeof(float));
  int*   top1  = (int*)alloc((size_t)B_ * sizeof(int));

  auto gemm = [&](const u16* A, int lda, const u16* W, const float* bias,
                  float* oF, u16* oB, int M, int N, int K) {
    dim3 grid(N / 256, M / 32);
    gemm_bf16_wmma<<<grid, 256, 0, stream>>>(A, lda, W, bias, oF, oB, M, N, K);
  };
  // One launch per LSTM timestep; h ping-pongs (hp -> hn).
  auto step = [&](const u16* X, int ldx, u16*& hp, u16*& hn, float* c,
                  const u16* W, const float* bsum, u16* ys, int ldy) {
    dim3 grid(G_ / 256, B_ / 32);   // 16 x 8
    lstm_step_fused<<<grid, 256, 0, stream>>>(X, ldx, hp, W, bsum, c, hn, ys, ldy);
    u16* t = hp; hp = hn; hn = t;   // hp now holds the latest h
  };

  const int BT = B_ * T_;

  // ---- Encoder FCs: e = L3(L2(L1(x))) (pure linear chain) ----
  gemm(x_bf, IN_, efc1_w, (const float*)d_in[2], nullptr, act0, BT, H_, IN_);
  gemm(act0, H_,  efc2_w, (const float*)d_in[4], nullptr, act1, BT, H_, H_);
  gemm(act1, H_,  efc3_w, (const float*)d_in[6], nullptr, act0, BT, H_, H_);

  // ---- zero LSTM state ----
  hipMemsetAsync(c1, 0, (size_t)B_ * H_ * sizeof(float), stream);
  hipMemsetAsync(c2, 0, (size_t)B_ * H_ * sizeof(float), stream);
  hipMemsetAsync(h1a, 0, (size_t)B_ * H_ * sizeof(u16), stream);
  hipMemsetAsync(h2a, 0, (size_t)B_ * H_ * sizeof(u16), stream);

  // ---- Encoder LSTM layer 1: reads act0 (B,T,H), ys -> act1 ----
  for (int t = 0; t < T_; ++t)
    step(act0 + (size_t)t * H_, T_ * H_, h1a, h1b, c1, el1_w, el1_b,
         act1 + (size_t)t * H_, T_ * H_);
  // ---- Encoder LSTM layer 2: reads act1, ys -> act0 ----
  for (int t = 0; t < T_; ++t)
    step(act1 + (size_t)t * H_, T_ * H_, h2a, h2b, c2, el2_w, el2_b,
         act0 + (size_t)t * H_, T_ * H_);

  // ---- Decoder ----
  hipMemsetAsync(d_out, 0, (size_t)out_size * sizeof(float), stream);

  for (int s = 0; s < DELTA_; ++s) {
    const u16* xs; int xlda;
    if (s == 0) { xs = act0 + (size_t)(T_ - 1) * H_; xlda = T_ * H_; }  // e[:, -1, :]
    else        { xs = emb;                           xlda = H_; }

    step(xs,  xlda, h1a, h1b, c1, dl1_w, dl1_b, nullptr, 0);   // dl1 cell
    step(h1a, H_,   h2a, h2b, c2, dl2_w, dl2_b, nullptr, 0);   // dl2 cell (input = new h1)

    // FC head -> logits over V=16384
    gemm(h2a,  H_, dfc1_w, (const float*)d_in[24], nullptr, bufA, B_, H_, H_);
    gemm(bufA, H_, dfc2_w, (const float*)d_in[26], nullptr, bufB, B_, H_, H_);
    gemm(bufB, H_, dfc3_w, (const float*)d_in[28], scores, nullptr, B_, V_, H_);

    beam_kernel<<<B_, 256, 0, stream>>>(scores, (float*)d_out, top1, s, (s == 0) ? KB_ : 1);

    if (s < DELTA_ - 1) {
      emb_kernel<<<(B_ * H_) / 256, 256, 0, stream>>>(
          top1,
          (const float*)d_in[29], (const float*)d_in[30],
          (const float*)d_in[31], (const float*)d_in[32],
          emb);
    }
  }
}
